// RNNModel_59021440582010
// MI455X (gfx1250) — matrix operand order repaired and checked by reading, not compiled, not run on hardware
//
#include <hip/hip_runtime.h>

// ---------------------------------------------------------------------------
// Types for CDNA5 WMMA (wave32): v_wmma_f32_16x16x32_bf16
// ---------------------------------------------------------------------------
typedef __attribute__((ext_vector_type(16))) __bf16 v16bf;
typedef __attribute__((ext_vector_type(8)))  __bf16 v8bf;
typedef __attribute__((ext_vector_type(8)))  float  v8f;

union ABFrag { v16bf v; __bf16 e[16]; };

// Model constants
#define NTOKEN 32000
#define NHID   1024
#define NLAY   2
#define SEQ    128
#define BATCH  32

// ---------------------------------------------------------------------------
// Fragment loaders (layouts per CDNA5 ISA §7.12.2, wave32)
//
// A (16x32 bf16, MxK): lane L (0..31): row = M = L&15, kg = L>>4.
//   elems[0..7]  = A[row, kbase + kg*8 + 0..7]
//   elems[8..15] = A[row, kbase + 16 + kg*8 + 0..7]
//
// B (32x16 bf16, KxN), sourced from a PRE-TRANSPOSED weight WT[n][k]:
//   lane L: col N = L&15, elems[0..7] = K (L>>4)*8 + 0..7, elems[8..15] = K 16 + (L>>4)*8 + 0..7
// ---------------------------------------------------------------------------
static __device__ inline v16bf load_a_frag(const __bf16* __restrict__ A, int lda,
                                           int row0, int kbase, int lane) {
  int row = row0 + (lane & 15);
  int kg  = (lane >> 4) * 8;
  const __bf16* p = A + (size_t)row * lda + kbase + kg;
  ABFrag r;
  *(v8bf*)(r.e)     = *(const v8bf*)(p);
  *(v8bf*)(r.e + 8) = *(const v8bf*)(p + 16);
  return r.v;
}

static __device__ inline v16bf load_b_frag(const __bf16* __restrict__ WT, int ldb,
                                           int n0, int kbase, int lane) {
  int n  = n0 + (lane & 15);
  int ko = (lane >> 4) * 8;
  const __bf16* p = WT + (size_t)n * ldb + kbase + ko;
  ABFrag r;
  *(v8bf*)(r.e)     = *(const v8bf*)(p);
  *(v8bf*)(r.e + 8) = *(const v8bf*)(p + 16);
  return r.v;
}

static __device__ inline v8f wmma_bf16(v16bf a, v16bf b, v8f c) {
  return __builtin_amdgcn_wmma_f32_16x16x32_bf16(
      /*neg_a=*/false, a, /*neg_b=*/false, b,
      /*c_mod=*/(short)0, c, /*reuse_a=*/false, /*reuse_b=*/false);
}

static __device__ inline float sigmoidf_(float x) {
  return 1.0f / (1.0f + __expf(-x));
}

// ---------------------------------------------------------------------------
// 1) Tiled transpose + fp32->bf16 downconvert: in[R][C] f32 -> out[C][R] bf16
//    (R, C multiples of 32). Block 256 threads = 32x8.
// ---------------------------------------------------------------------------
__global__ __launch_bounds__(256) void transpose_bf16_kernel(
    const float* __restrict__ in, __bf16* __restrict__ out, int R, int C) {
  __shared__ float tile[32][33];
  int tx = threadIdx.x & 31;
  int ty = threadIdx.x >> 5;
  int c0 = blockIdx.x * 32;
  int r0 = blockIdx.y * 32;
#pragma unroll
  for (int i = 0; i < 32; i += 8)
    tile[ty + i][tx] = in[(size_t)(r0 + ty + i) * C + (c0 + tx)];
  __syncthreads();
#pragma unroll
  for (int i = 0; i < 32; i += 8)
    out[(size_t)(c0 + ty + i) * R + (r0 + tx)] = (__bf16)tile[tx][ty + i];
}

// ---------------------------------------------------------------------------
// 2) Embedding gather -> bf16 rows. One block per (s,b) row.
// ---------------------------------------------------------------------------
__global__ __launch_bounds__(256) void embed_kernel(
    const int* __restrict__ tokens, const float* __restrict__ table,
    __bf16* __restrict__ out) {
  int row = blockIdx.x;  // row = s*BATCH + b
  int tok = tokens[row];
  const float* src = table + (size_t)tok * NHID;
  __bf16* dst = out + (size_t)row * NHID;
  for (int i = threadIdx.x; i < NHID; i += 256) dst[i] = (__bf16)src[i];
}

// ---------------------------------------------------------------------------
// 3) State init from h0/c0 inputs (deterministic per call)
// ---------------------------------------------------------------------------
__global__ __launch_bounds__(256) void init_state_kernel(
    const float* __restrict__ h0, const float* __restrict__ c0,
    float* __restrict__ Hf, float* __restrict__ Cf, __bf16* __restrict__ Hb) {
  int i = blockIdx.x * 256 + threadIdx.x;
  if (i < NLAY * BATCH * NHID) {
    float h = h0[i];
    Hf[i] = h;
    Hb[i] = (__bf16)h;
    Cf[i] = c0[i];
  }
}

// ---------------------------------------------------------------------------
// 4) One LSTM layer step.
//    gates[32,4096] = [x | h] @ W  with W pre-transposed bf16 WT[4096][2048].
//    Block computes ALL FOUR gates for a 32-wide hidden slice so the cell
//    update fuses in-block via LDS. 8 waves: wave w -> gate g=w>>1,
//    half=w&1; each wave does both M-tiles (M=32). Grid: 1024/32 = 32 blocks.
//    K-loop is software-pipelined (next fragments in flight during WMMA).
// ---------------------------------------------------------------------------
__global__ __launch_bounds__(256) void lstm_step_kernel(
    const __bf16* __restrict__ xa,    // [32,1024] layer input (bf16)
    const __bf16* __restrict__ ha,    // [32,1024] prev hidden (bf16)
    const __bf16* __restrict__ WT,    // [4096,2048] bf16 (transposed weights)
    const float*  __restrict__ bias,  // [4096]
    float*  __restrict__ Cst,         // [32,1024] cell state (in/out)
    float*  __restrict__ Hf,          // [32,1024] hidden f32 out
    __bf16* __restrict__ Hb,          // [32,1024] hidden bf16 out
    __bf16* __restrict__ OutRow)      // optional [32,1024] sequence-output row
{
  __shared__ float gates[4][BATCH][32];
  const int tid  = threadIdx.x;
  const int wave = tid >> 5;
  const int lane = tid & 31;
  const int g    = wave >> 1;
  const int half = wave & 1;
  const int h0   = blockIdx.x * 32;
  const int n0   = g * NHID + h0 + half * 16;

  v8f acc0 = {};
  v8f acc1 = {};

  // Pipelined K loop over K = 0..2047 (first half from xa, second from ha).
  v16bf a0 = load_a_frag(xa, NHID, 0,  0, lane);
  v16bf a1 = load_a_frag(xa, NHID, 16, 0, lane);
  v16bf b  = load_b_frag(WT, 2 * NHID, n0, 0, lane);
#pragma unroll 4
  for (int kb = 32; kb < 2 * NHID; kb += 32) {
    const __bf16* Ap = (kb < NHID) ? xa : ha;
    const int ka = (kb < NHID) ? kb : (kb - NHID);
    v16bf na0 = load_a_frag(Ap, NHID, 0,  ka, lane);
    v16bf na1 = load_a_frag(Ap, NHID, 16, ka, lane);
    v16bf nb  = load_b_frag(WT, 2 * NHID, n0, kb, lane);
    acc0 = wmma_bf16(a0, b, acc0);
    acc1 = wmma_bf16(a1, b, acc1);
    a0 = na0; a1 = na1; b = nb;
  }
  acc0 = wmma_bf16(a0, b, acc0);
  acc1 = wmma_bf16(a1, b, acc1);

  // D layout: lanes 0-15 -> N=lane, M=vgpr r; lanes 16-31 -> N=lane-16, M=8+r
  const int col   = half * 16 + (lane & 15);
  const int rbase = (lane >> 4) * 8;
#pragma unroll
  for (int r = 0; r < 8; r++) {
    gates[g][rbase + r][col]      = acc0[r];
    gates[g][16 + rbase + r][col] = acc1[r];
  }
  __syncthreads();

  // Fused cell update for this hidden slice (gate order: i, j, f, o)
  for (int idx = tid; idx < BATCH * 32; idx += 256) {
    int b2 = idx >> 5;
    int hh = idx & 31;
    int hg = h0 + hh;
    float gi = gates[0][b2][hh] + bias[hg];
    float gj = gates[1][b2][hh] + bias[NHID + hg];
    float gf = gates[2][b2][hh] + bias[2 * NHID + hg];
    float go = gates[3][b2][hh] + bias[3 * NHID + hg];
    float cold = Cst[b2 * NHID + hg];
    float cnew = cold * sigmoidf_(gf) + sigmoidf_(gi) * tanhf(gj);
    float hnew = tanhf(cnew) * sigmoidf_(go);
    Cst[b2 * NHID + hg] = cnew;
    Hf[b2 * NHID + hg]  = hnew;
    __bf16 hb = (__bf16)hnew;
    Hb[b2 * NHID + hg] = hb;
    if (OutRow) OutRow[b2 * NHID + hg] = hb;
  }
}

// ---------------------------------------------------------------------------
// 5) Decoder GEMM: [4096,1024]bf16 @ decWT[32000][1024]bf16 -> f32 + bias.
//    Block = 8 waves, tile M=32 x N=64 (2x4 waves). Grid (128, 500).
//    Pipelined K loop; result staged in LDS for fully coalesced f32 stores
//    (stores dominate: 524 MB of logits => output-bandwidth bound).
// ---------------------------------------------------------------------------
__global__ __launch_bounds__(256) void decoder_kernel(
    const __bf16* __restrict__ X,    // [SEQ*BATCH, NHID]
    const __bf16* __restrict__ WT,   // [NTOKEN, NHID]
    const float*  __restrict__ bias, // [NTOKEN]
    float* __restrict__ Out)         // [SEQ*BATCH, NTOKEN]
{
  __shared__ float otile[32][65];    // 32 x 64 block tile (+pad)
  const int tid  = threadIdx.x;
  const int wave = tid >> 5;
  const int lane = tid & 31;
  const int mt   = wave & 1;
  const int nt   = wave >> 1;
  const int mB   = blockIdx.x * 32;
  const int nB   = blockIdx.y * 64;
  const int m0   = mB + mt * 16;
  const int n0   = nB + nt * 16;

  v8f acc = {};
  v16bf a = load_a_frag(X, NHID, m0, 0, lane);
  v16bf b = load_b_frag(WT, NHID, n0, 0, lane);
#pragma unroll 4
  for (int kb = 32; kb < NHID; kb += 32) {
    v16bf na = load_a_frag(X, NHID, m0, kb, lane);
    v16bf nb = load_b_frag(WT, NHID, n0, kb, lane);
    acc = wmma_bf16(a, b, acc);
    a = na; b = nb;
  }
  acc = wmma_bf16(a, b, acc);

  // Stage D tile in LDS: lanes 0-15 -> N=lane, M=r; lanes 16-31 -> N=lane-16, M=8+r
  const int col   = nt * 16 + (lane & 15);
  const int rbase = mt * 16 + (lane >> 4) * 8;
#pragma unroll
  for (int r = 0; r < 8; r++) otile[rbase + r][col] = acc[r];
  __syncthreads();

  // Coalesced writeback: each 64-thread group writes one contiguous 256B row.
#pragma unroll
  for (int i = 0; i < 8; i++) {
    int idx = i * 256 + tid;
    int row = idx >> 6;
    int c   = idx & 63;
    Out[(size_t)(mB + row) * NTOKEN + nB + c] = otile[row][c] + bias[nB + c];
  }
}

// ---------------------------------------------------------------------------
// 6) Copy final (h_last, c_last) to d_out tail
// ---------------------------------------------------------------------------
__global__ __launch_bounds__(256) void finalize_kernel(
    const float* __restrict__ Hf, const float* __restrict__ Cf,
    float* __restrict__ out) {
  int i = blockIdx.x * 256 + threadIdx.x;
  const size_t dec = (size_t)SEQ * BATCH * NTOKEN;
  const int n = NLAY * BATCH * NHID;
  if (i < n) {
    out[dec + i]     = Hf[i];
    out[dec + n + i] = Cf[i];
  }
}

// ---------------------------------------------------------------------------
// Host orchestration
// ---------------------------------------------------------------------------
extern "C" void kernel_launch(void* const* d_in, const int* in_sizes, int n_in,
                              void* d_out, int out_size, void* d_ws, size_t ws_size,
                              hipStream_t stream) {
  (void)in_sizes; (void)n_in; (void)out_size; (void)ws_size;
  const int*   tokens    = (const int*)  d_in[0];
  const float* h0        = (const float*)d_in[1];
  const float* c0        = (const float*)d_in[2];
  const float* emb_table = (const float*)d_in[3];
  const float* lstm_W    = (const float*)d_in[4];  // [2][2048][4096]
  const float* lstm_b    = (const float*)d_in[5];  // [2][4096]
  const float* dec_W     = (const float*)d_in[6];  // [1024][32000]
  const float* dec_b     = (const float*)d_in[7];  // [32000]
  float* out = (float*)d_out;

  // Workspace layout (all offsets 256B-aligned)
  char* ws = (char*)d_ws;
  __bf16* embT    = (__bf16*)ws;                 ws += (size_t)SEQ * BATCH * NHID * 2;         // 8 MB
  __bf16* seqOut  = (__bf16*)ws;                 ws += (size_t)SEQ * BATCH * NHID * 2;         // 8 MB
  __bf16* lstmWT  = (__bf16*)ws;                 ws += (size_t)NLAY * 4 * NHID * 2 * NHID * 2; // 33.5 MB
  __bf16* decWT   = (__bf16*)ws;                 ws += (size_t)NTOKEN * NHID * 2;              // 65.5 MB
  __bf16* Hb      = (__bf16*)ws;                 ws += (size_t)NLAY * BATCH * NHID * 2;        // 128 KB
  float*  Hf      = (float*)ws;                  ws += (size_t)NLAY * BATCH * NHID * 4;        // 256 KB
  float*  Cf      = (float*)ws;                  /* ws += 256 KB */

  const size_t Wl = (size_t)2 * NHID * 4 * NHID;  // elems per layer of lstm_W / lstmWT

  // Weight transposes + downconvert (in [R][C] f32 -> out [C][R] bf16)
  transpose_bf16_kernel<<<dim3(4 * NHID / 32, 2 * NHID / 32), 256, 0, stream>>>(
      lstm_W, lstmWT, 2 * NHID, 4 * NHID);
  transpose_bf16_kernel<<<dim3(4 * NHID / 32, 2 * NHID / 32), 256, 0, stream>>>(
      lstm_W + Wl, lstmWT + Wl, 2 * NHID, 4 * NHID);
  transpose_bf16_kernel<<<dim3(NTOKEN / 32, NHID / 32), 256, 0, stream>>>(
      dec_W, decWT, NHID, NTOKEN);

  // Embedding gather
  embed_kernel<<<SEQ * BATCH, 256, 0, stream>>>(tokens, emb_table, embT);

  // h/c state init from inputs
  init_state_kernel<<<(NLAY * BATCH * NHID + 255) / 256, 256, 0, stream>>>(
      h0, c0, Hf, Cf, Hb);

  // Sequential LSTM scan: 128 steps x 2 layers
  const size_t lst = (size_t)BATCH * NHID;  // per-layer state stride
  for (int t = 0; t < SEQ; ++t) {
    const __bf16* x0 = embT + (size_t)t * BATCH * NHID;
    // layer 0
    lstm_step_kernel<<<NHID / 32, 256, 0, stream>>>(
        x0, Hb, lstmWT, lstm_b, Cf, Hf, Hb, (__bf16*)nullptr);
    // layer 1 (input = layer 0's fresh hidden); also emits sequence output row
    lstm_step_kernel<<<NHID / 32, 256, 0, stream>>>(
        Hb, Hb + lst, lstmWT + Wl, lstm_b + 4 * NHID,
        Cf + lst, Hf + lst, Hb + lst,
        seqOut + (size_t)t * BATCH * NHID);
  }

  // Decoder GEMM: [4096,1024] @ [1024,32000] + bias -> d_out
  decoder_kernel<<<dim3(SEQ * BATCH / 32, NTOKEN / 64), 256, 0, stream>>>(
      seqOut, decWT, dec_b, out);

  // Tail: h_last, c_last
  finalize_kernel<<<(NLAY * BATCH * NHID + 255) / 256, 256, 0, stream>>>(Hf, Cf, out);
}
